// TambaMambaEncoder_10024453669344
// MI455X (gfx1250) — matrix operand order repaired and checked by reading, not compiled, not run on hardware
//
#include <hip/hip_runtime.h>
#include <hip/hip_fp16.h>

// ---------------------------------------------------------------------------
// Mamba encoder (4 layers) for MI455X / gfx1250.
// GEMMs run on v_wmma_f32_16x16x32_f16 (f16 in, f32 accumulate).
//   - main GEMMs: 16x64 output tile per wave (1 A fragment feeds 4
//     independent WMMAs -> 4x less A traffic, deeper XDL pipelining)
//   - x_proj (N=96): 16x16-per-wave fallback
// Selective scan: 1 thread per (batch, channel), 16-wide state in VGPRs,
// B/C vectors staged in LDS (shared across all channels of a batch).
// ---------------------------------------------------------------------------

typedef __attribute__((ext_vector_type(16))) _Float16 v16h;
typedef __attribute__((ext_vector_type(8)))  _Float16 v8h;
typedef __attribute__((ext_vector_type(8)))  float    v8f;

#define D_MODEL  1024
#define N_LAYERS 4
#define INTER    2048
#define DT_RANK  64
#define D_STATE  16
#define D_CONV   4
#define B_SZ     4
#define SEQ      1024
#define NTOK     (B_SZ * SEQ)             // 4096 tokens
#define SSM_W    (DT_RANK + 2 * D_STATE)  // 96
#define EPS      1e-5f

// ---------------------------------------------------------------------------
// WMMA fragment loaders (layouts per CDNA5 ISA 7.12.2):
//   A (16x32): lane l holds row (l&15); K chunks [8*(l>>4),+8) and
//              [16+8*(l>>4),+8)  -> two b128 loads
//   B (32x16): lane l holds col (l&15); K chunks [8*(l>>4),+8) and
//              [16+8*(l>>4),+8)  -> two b128 loads
//   C (16x16): VGPR r -> row r + 8*(l>>4), col (l&15)
// ---------------------------------------------------------------------------
__device__ __forceinline__ v16h load_a_frag(const _Float16* p, int hi) {
  v8h a0 = *(const v8h*)(p + hi * 8);
  v8h a1 = *(const v8h*)(p + 16 + hi * 8);
  return __builtin_shufflevector(a0, a1, 0, 1, 2, 3, 4, 5, 6, 7,
                                 8, 9, 10, 11, 12, 13, 14, 15);
}
__device__ __forceinline__ v16h load_b_frag(const _Float16* p, int hi) {
  v8h b0 = *(const v8h*)(p + hi * 8);
  v8h b1 = *(const v8h*)(p + 16 + hi * 8);
  return __builtin_shufflevector(b0, b1, 0, 1, 2, 3, 4, 5, 6, 7,
                                 8, 9, 10, 11, 12, 13, 14, 15);
}
__device__ __forceinline__ v8f wmma_step(v16h a, v16h b, v8f c) {
  return __builtin_amdgcn_wmma_f32_16x16x32_f16(
      /*neg_a=*/false, a, /*neg_b=*/false, b,
      /*c_mod=*/(short)0, c, /*reuse_a=*/false, /*reuse_b=*/false);
}

// ---------------------------------------------------------------------------
// f32 -> f16 conversion (weights, once per launch)
// ---------------------------------------------------------------------------
__global__ void cvt_f32_f16(const float* __restrict__ src,
                            _Float16* __restrict__ dst, int n) {
  int i = blockIdx.x * blockDim.x + threadIdx.x;
  if (i < n) dst[i] = (_Float16)src[i];
}

// ---------------------------------------------------------------------------
// RMSNorm over D_MODEL per token. OUT_HALF=true -> f16 (GEMM input),
// else f32 (final output).
// ---------------------------------------------------------------------------
template <bool OUT_HALF>
__global__ void rmsnorm_kernel(const float* __restrict__ x,
                               const float* __restrict__ w,
                               float* __restrict__ of32,
                               _Float16* __restrict__ of16) {
  __shared__ float red[256];
  const int row = blockIdx.x;
  const float* xr = x + (size_t)row * D_MODEL;
  float s = 0.f;
#pragma unroll
  for (int j = 0; j < D_MODEL / 256; ++j) {
    float v = xr[threadIdx.x + j * 256];
    s += v * v;
  }
  red[threadIdx.x] = s;
  __syncthreads();
  for (int off = 128; off > 0; off >>= 1) {
    if (threadIdx.x < off) red[threadIdx.x] += red[threadIdx.x + off];
    __syncthreads();
  }
  const float scale = rsqrtf(red[0] * (1.0f / D_MODEL) + EPS);
#pragma unroll
  for (int j = 0; j < D_MODEL / 256; ++j) {
    int d = threadIdx.x + j * 256;
    float v = xr[d] * scale * w[d];
    if (OUT_HALF) of16[(size_t)row * D_MODEL + d] = (_Float16)v;
    else          of32[(size_t)row * D_MODEL + d] = v;
  }
}

// ---------------------------------------------------------------------------
// C[M,N] = A[M,K] * W[N,K]^T ; 16x64 tile per wave (N % 64 == 0).
// One A fragment feeds 4 independent WMMAs per K-step.
// ---------------------------------------------------------------------------
__global__ void wmma_gemm_xwt_x4(const _Float16* __restrict__ A,
                                 const _Float16* __restrict__ W,
                                 float* __restrict__ C,
                                 int M, int N, int K) {
  const int lane = threadIdx.x & 31;
  const int wave = blockIdx.x * (blockDim.x >> 5) + (threadIdx.x >> 5);
  const int ncg  = N >> 6;                 // 64-wide column groups
  const int tm   = wave / ncg;
  const int tn0  = (wave % ncg) << 2;      // first 16-wide column tile
  if (tm * 16 >= M) return;                // wave-uniform: EXEC stays all-1s

  const int r  = lane & 15;
  const int hi = lane >> 4;
  const _Float16* ap  = A + (size_t)(tm * 16 + r) * K;
  const _Float16* wp0 = W + (size_t)(tn0 * 16 + r) * K;
  const _Float16* wp1 = wp0 + (size_t)16 * K;
  const _Float16* wp2 = wp0 + (size_t)32 * K;
  const _Float16* wp3 = wp0 + (size_t)48 * K;

  v8f c0 = {}, c1 = {}, c2 = {}, c3 = {};
  for (int k0 = 0; k0 < K; k0 += 32) {
    __builtin_prefetch(ap + k0 + 512, 0, 0);   // global_prefetch_b8
    v16h a = load_a_frag(ap + k0, hi);
    c0 = wmma_step(a, load_b_frag(wp0 + k0, hi), c0);
    c1 = wmma_step(a, load_b_frag(wp1 + k0, hi), c1);
    c2 = wmma_step(a, load_b_frag(wp2 + k0, hi), c2);
    c3 = wmma_step(a, load_b_frag(wp3 + k0, hi), c3);
  }

  float* cp = C + (size_t)(tm * 16 + hi * 8) * N + tn0 * 16 + r;
#pragma unroll
  for (int i = 0; i < 8; ++i) {
    cp[(size_t)i * N + 0]  = c0[i];
    cp[(size_t)i * N + 16] = c1[i];
    cp[(size_t)i * N + 32] = c2[i];
    cp[(size_t)i * N + 48] = c3[i];
  }
}

// ---------------------------------------------------------------------------
// Fallback: one 16x16 tile per wave (any N % 16 == 0). Used for x_proj.
// ---------------------------------------------------------------------------
__global__ void wmma_gemm_xwt(const _Float16* __restrict__ A,
                              const _Float16* __restrict__ W,
                              float* __restrict__ C,
                              int M, int N, int K) {
  const int lane = threadIdx.x & 31;
  const int wave = blockIdx.x * (blockDim.x >> 5) + (threadIdx.x >> 5);
  const int ntn  = N >> 4;
  const int tm   = wave / ntn;
  const int tn   = wave % ntn;
  if (tm * 16 >= M) return;

  const int r  = lane & 15;
  const int hi = lane >> 4;
  const _Float16* ap = A + (size_t)(tm * 16 + r) * K;
  const _Float16* wp = W + (size_t)(tn * 16 + r) * K;

  v8f acc = {};
  for (int k0 = 0; k0 < K; k0 += 32) {
    __builtin_prefetch(ap + k0 + 512, 0, 0);
    v16h a = load_a_frag(ap + k0, hi);
    v16h b = load_b_frag(wp + k0, hi);
    acc = wmma_step(a, b, acc);
  }

  float* cp = C + (size_t)(tm * 16 + hi * 8) * N + tn * 16 + r;
#pragma unroll
  for (int i = 0; i < 8; ++i) cp[(size_t)i * N] = acc[i];
}

// ---------------------------------------------------------------------------
// Causal depthwise conv (width 4) along time + bias + SiLU.
// hs = proj[:, :INTER]. Emits u in f32 (scan) and f16 (x_proj GEMM).
// ---------------------------------------------------------------------------
__global__ void conv_silu_kernel(const float* __restrict__ proj,  // [NTOK, 2I]
                                 const float* __restrict__ cw,    // [I, 4]
                                 const float* __restrict__ cb,    // [I]
                                 float* __restrict__ u,           // [NTOK, I]
                                 _Float16* __restrict__ u16) {    // [NTOK, I]
  int idx = blockIdx.x * blockDim.x + threadIdx.x;
  if (idx >= NTOK * INTER) return;
  const int i = idx % INTER;
  const int m = idx / INTER;       // m = b*SEQ + t
  const int t = m % SEQ;
  float s = cb[i];
#pragma unroll
  for (int k = 0; k < D_CONV; ++k) {
    int tt = t - (D_CONV - 1) + k;
    if (tt >= 0)
      s += cw[i * D_CONV + k] *
           proj[(size_t)(m - (D_CONV - 1) + k) * (2 * INTER) + i];
  }
  float sv = s / (1.f + __expf(-s));  // SiLU
  u[idx]   = sv;
  u16[idx] = (_Float16)sv;
}

// ---------------------------------------------------------------------------
// Slice dt columns of ssm ([NTOK,96] -> first 64) to f16 for dt_proj GEMM.
// ---------------------------------------------------------------------------
__global__ void slice_dt_kernel(const float* __restrict__ ssm,
                                _Float16* __restrict__ dt16) {
  int idx = blockIdx.x * blockDim.x + threadIdx.x;
  if (idx >= NTOK * DT_RANK) return;
  int m = idx / DT_RANK, k = idx % DT_RANK;
  dt16[idx] = (_Float16)ssm[(size_t)m * SSM_W + k];
}

// ---------------------------------------------------------------------------
// Selective scan: 1 thread per (b, channel i). State[16] in VGPRs.
// B/C (per-(b,t), shared by all channels) staged in LDS per 64-step chunk.
// Fuses: softplus(dt), dA=exp(dt*A), state update, y = state.C + u*D,
// gate SiLU multiply, f16 conversion for out_proj.
// ---------------------------------------------------------------------------
#define TC 64
__global__ void scan_kernel(const float* __restrict__ ssm,    // [NTOK, 96]
                            const float* __restrict__ dtl,    // [NTOK, I]
                            const float* __restrict__ u,      // [NTOK, I]
                            const float* __restrict__ proj,   // [NTOK, 2I]
                            const float* __restrict__ A_log,  // [I, 16]
                            const float* __restrict__ Dp,     // [I]
                            const float* __restrict__ dt_b,   // [I]
                            _Float16* __restrict__ y16) {     // [NTOK, I]
  __shared__ float Bsh[TC][D_STATE];
  __shared__ float Csh[TC][D_STATE];
  const int i = blockIdx.x * blockDim.x + threadIdx.x;  // channel
  const int b = blockIdx.y;

  float Arow[D_STATE];
#pragma unroll
  for (int n = 0; n < D_STATE; ++n)
    Arow[n] = -__expf(A_log[(size_t)i * D_STATE + n]);
  const float Dv = Dp[i], dtb = dt_b[i];

  float st[D_STATE];
#pragma unroll
  for (int n = 0; n < D_STATE; ++n) st[n] = 0.f;

  for (int t0 = 0; t0 < SEQ; t0 += TC) {
    __syncthreads();
    for (int e = threadIdx.x; e < TC * 2 * D_STATE; e += blockDim.x) {
      int tl = e >> 5, j = e & 31;
      float v = ssm[(size_t)(b * SEQ + t0 + tl) * SSM_W + DT_RANK + j];
      if (j < D_STATE) Bsh[tl][j] = v;
      else             Csh[tl][j - D_STATE] = v;
    }
    __syncthreads();

    for (int tl = 0; tl < TC; ++tl) {
      const size_t m = (size_t)(b * SEQ + t0 + tl);
      float dtr = dtl[m * INTER + i] + dtb;
      float dt  = (dtr > 20.f) ? dtr : log1pf(__expf(dtr));  // softplus
      float uv  = u[m * INTER + i];
      float du  = dt * uv;
      float y   = 0.f;
#pragma unroll
      for (int n = 0; n < D_STATE; ++n) {
        float dA = __expf(dt * Arow[n]);
        st[n] = dA * st[n] + du * Bsh[tl][n];
        y += st[n] * Csh[tl][n];
      }
      y += uv * Dv;
      float g = proj[m * (2 * INTER) + INTER + i];
      y *= g / (1.f + __expf(-g));                           // * silu(gate)
      y16[m * INTER + i] = (_Float16)y;
    }
  }
}

// ---------------------------------------------------------------------------
// h = a + tmp (residual). Layer 0 uses a = x (input), later layers a = h.
// ---------------------------------------------------------------------------
__global__ void add_kernel(const float* __restrict__ a,
                           const float* __restrict__ tmp,
                           float* __restrict__ h, int n) {
  int idx = blockIdx.x * blockDim.x + threadIdx.x;
  if (idx < n) h[idx] = a[idx] + tmp[idx];
}

// ---------------------------------------------------------------------------
// Host-side launch orchestration (all on `stream`, no allocations).
// ---------------------------------------------------------------------------
static inline int ceil_div(long a, long b) { return (int)((a + b - 1) / b); }

extern "C" void kernel_launch(void* const* d_in, const int* in_sizes, int n_in,
                              void* d_out, int out_size, void* d_ws,
                              size_t ws_size, hipStream_t stream) {
  (void)in_sizes; (void)n_in; (void)out_size; (void)ws_size;

  const float* x      = (const float*)d_in[0];
  const float* in_w   = (const float*)d_in[1];
  const float* conv_w = (const float*)d_in[2];
  const float* conv_b = (const float*)d_in[3];
  const float* x_w    = (const float*)d_in[4];
  const float* dt_w   = (const float*)d_in[5];
  const float* dt_b   = (const float*)d_in[6];
  const float* A_log  = (const float*)d_in[7];
  const float* Dp     = (const float*)d_in[8];
  const float* out_w  = (const float*)d_in[9];
  const float* norm_w = (const float*)d_in[10];
  const float* norm_f = (const float*)d_in[11];

  // Workspace bump allocator (256B aligned).
  char* p = (char*)d_ws;
  auto alloc = [&](size_t bytes) -> void* {
    void* r = (void*)p;
    p += (bytes + 255) & ~(size_t)255;
    return r;
  };

  float*    h      = (float*)alloc((size_t)NTOK * D_MODEL * 4);
  _Float16* inw16  = (_Float16*)alloc((size_t)N_LAYERS * 2 * INTER * D_MODEL * 2);
  _Float16* xw16   = (_Float16*)alloc((size_t)N_LAYERS * SSM_W * INTER * 2);
  _Float16* dtw16  = (_Float16*)alloc((size_t)N_LAYERS * INTER * DT_RANK * 2);
  _Float16* outw16 = (_Float16*)alloc((size_t)N_LAYERS * D_MODEL * INTER * 2);
  _Float16* xn16   = (_Float16*)alloc((size_t)NTOK * D_MODEL * 2);
  float*    proj   = (float*)alloc((size_t)NTOK * 2 * INTER * 4);
  float*    u      = (float*)alloc((size_t)NTOK * INTER * 4);
  _Float16* u16    = (_Float16*)alloc((size_t)NTOK * INTER * 2);
  float*    ssm    = (float*)alloc((size_t)NTOK * SSM_W * 4);
  _Float16* dt16   = (_Float16*)alloc((size_t)NTOK * DT_RANK * 2);
  float*    dtl    = (float*)alloc((size_t)NTOK * INTER * 4);
  _Float16* y16    = (_Float16*)alloc((size_t)NTOK * INTER * 2);
  float*    tmp    = (float*)alloc((size_t)NTOK * D_MODEL * 4);

  // ---- convert all weights to f16 once (deterministic, re-done each call)
  {
    long n;
    n = (long)N_LAYERS * 2 * INTER * D_MODEL;
    cvt_f32_f16<<<ceil_div(n, 256), 256, 0, stream>>>(in_w, inw16, (int)n);
    n = (long)N_LAYERS * SSM_W * INTER;
    cvt_f32_f16<<<ceil_div(n, 256), 256, 0, stream>>>(x_w, xw16, (int)n);
    n = (long)N_LAYERS * INTER * DT_RANK;
    cvt_f32_f16<<<ceil_div(n, 256), 256, 0, stream>>>(dt_w, dtw16, (int)n);
    n = (long)N_LAYERS * D_MODEL * INTER;
    cvt_f32_f16<<<ceil_div(n, 256), 256, 0, stream>>>(out_w, outw16, (int)n);
  }

  const int WPB = 8;  // waves per block (256 threads)
  auto blocks_x4 = [&](int M, int N) {  // 16x64 tiles
    long tiles = (long)(M / 16) * (N / 64);
    return ceil_div(tiles, WPB);
  };
  auto blocks_x1 = [&](int M, int N) {  // 16x16 tiles
    long tiles = (long)(M / 16) * (N / 16);
    return ceil_div(tiles, WPB);
  };

  for (int l = 0; l < N_LAYERS; ++l) {
    const float* hin = (l == 0) ? x : h;

    // 1) RMSNorm -> f16
    rmsnorm_kernel<true><<<NTOK, 256, 0, stream>>>(
        hin, norm_w + (size_t)l * D_MODEL, (float*)nullptr, xn16);

    // 2) in_proj: [4096,1024] x [4096,1024]^T -> [4096,4096]
    wmma_gemm_xwt_x4<<<blocks_x4(NTOK, 2 * INTER), 32 * WPB, 0, stream>>>(
        xn16, inw16 + (size_t)l * 2 * INTER * D_MODEL, proj,
        NTOK, 2 * INTER, D_MODEL);

    // 3) depthwise conv + SiLU -> u (f32 + f16)
    conv_silu_kernel<<<ceil_div((long)NTOK * INTER, 256), 256, 0, stream>>>(
        proj, conv_w + (size_t)l * INTER * D_CONV, conv_b + (size_t)l * INTER,
        u, u16);

    // 4) x_proj: [4096,2048] x [96,2048]^T -> [4096,96]   (N=96 -> x1 kernel)
    wmma_gemm_xwt<<<blocks_x1(NTOK, SSM_W), 32 * WPB, 0, stream>>>(
        u16, xw16 + (size_t)l * SSM_W * INTER, ssm, NTOK, SSM_W, INTER);

    // 5) slice dt columns -> f16
    slice_dt_kernel<<<ceil_div((long)NTOK * DT_RANK, 256), 256, 0, stream>>>(
        ssm, dt16);

    // 6) dt_proj: [4096,64] x [2048,64]^T -> [4096,2048]
    wmma_gemm_xwt_x4<<<blocks_x4(NTOK, INTER), 32 * WPB, 0, stream>>>(
        dt16, dtw16 + (size_t)l * INTER * DT_RANK, dtl, NTOK, INTER, DT_RANK);

    // 7) selective scan (fused softplus, D-skip, gate SiLU) -> y16
    scan_kernel<<<dim3(INTER / 256, B_SZ), 256, 0, stream>>>(
        ssm, dtl, u, proj, A_log + (size_t)l * INTER * D_STATE,
        Dp + (size_t)l * INTER, dt_b + (size_t)l * INTER, y16);

    // 8) out_proj: [4096,2048] x [1024,2048]^T -> [4096,1024]
    wmma_gemm_xwt_x4<<<blocks_x4(NTOK, D_MODEL), 32 * WPB, 0, stream>>>(
        y16, outw16 + (size_t)l * D_MODEL * INTER, tmp, NTOK, D_MODEL, INTER);

    // 9) residual
    add_kernel<<<ceil_div((long)NTOK * D_MODEL, 256), 256, 0, stream>>>(
        hin, tmp, h, NTOK * D_MODEL);
  }

  // Final RMSNorm -> d_out (f32)
  rmsnorm_kernel<false><<<NTOK, 256, 0, stream>>>(
      h, norm_f, (float*)d_out, (_Float16*)nullptr);
}
